// SimplifiedMultiHeadAttention_45595372814916
// MI455X (gfx1250) — compile-verified, matrix operand order checked by reading, not run on hardware
//
#include <hip/hip_runtime.h>
#include <hip/hip_bf16.h>

#define B_SZ   2
#define S_LEN  2048
#define D_IN   1024
#define N_H    16
#define H_D    64
#define M_TILE 128
#define K_TILE 32

typedef __attribute__((ext_vector_type(16))) __bf16 v16bf;
typedef __attribute__((ext_vector_type(8)))  float  v8f;

union BFrag {
    v16bf v;
    uint4 q[2];
};

// Hardware f32 -> bf16 convert (single v_cvt; clang fuses pairs to v_cvt_pk)
__device__ __forceinline__ unsigned short f32_to_bf16(float f) {
    union { __bf16 b; unsigned short u; } c;
    c.b = (__bf16)f;
    return c.u;
}
__device__ __forceinline__ unsigned int pack_bf16x2(float lo, float hi) {
    return (unsigned int)f32_to_bf16(lo) | ((unsigned int)f32_to_bf16(hi) << 16);
}

// ---------------------------------------------------------------------------
// Kernel 1: fused Q/K/V projection (bf16 WMMA, f32 accumulate, bias add)
// q,k stored [B*H, S, 64] bf16 ; v stored transposed [B*H, 64, S] bf16
// ---------------------------------------------------------------------------
__global__ __launch_bounds__(256) void qkv_proj_kernel(
    const float* __restrict__ x,
    const float* __restrict__ Wq, const float* __restrict__ bq,
    const float* __restrict__ Wk, const float* __restrict__ bk,
    const float* __restrict__ Wv, const float* __restrict__ bv,
    unsigned short* __restrict__ qws,
    unsigned short* __restrict__ kws,
    unsigned short* __restrict__ vtws)
{
    __shared__ __align__(16) unsigned short xs [M_TILE * K_TILE]; // [128][32]
    __shared__ __align__(16) unsigned short wqs[H_D * K_TILE];    // [64][32] (transposed)
    __shared__ __align__(16) unsigned short wks[H_D * K_TILE];
    __shared__ __align__(16) unsigned short wvs[H_D * K_TILE];

    const int h     = blockIdx.x % N_H;
    const int mt    = blockIdx.x / N_H;
    const int rbase = mt * M_TILE;
    const int t     = threadIdx.x;
    const int lane  = t & 31;
    const int w     = t >> 5;
    const int mloc  = lane & 15;
    const int kb    = (lane < 16) ? 0 : 8;   // K sub-block per ISA bf16 layout
    const int qsel  = kb >> 3;               // 0 or 1 (uint4 index)

    v8f acc[3][4];
#pragma unroll
    for (int p = 0; p < 3; p++)
#pragma unroll
        for (int n = 0; n < 4; n++)
#pragma unroll
            for (int j = 0; j < 8; j++) acc[p][n][j] = 0.0f;

    unsigned int* xs32 = (unsigned int*)xs;

    for (int kk = 0; kk < D_IN; kk += K_TILE) {
        // stage x tile (128x32 f32 -> bf16, pair-packed b32 stores)
#pragma unroll
        for (int i = 0; i < 8; i++) {
            int e2 = t + 256 * i;                // pair index, 0..2047
            int r = e2 >> 4, cp = e2 & 15;       // row, col-pair
            const float2 xv = *(const float2*)&x[(size_t)(rbase + r) * D_IN + kk + 2 * cp];
            xs32[e2] = pack_bf16x2(xv.x, xv.y);
        }
        // stage W tiles transposed ([n][k], pairs packed along k)
#pragma unroll
        for (int p = 0; p < 3; p++) {
            const float* W = (p == 0) ? (Wq + (size_t)h * D_IN * H_D)
                           : (p == 1) ? (Wk + (size_t)h * D_IN * H_D)
                                      : (Wv + (size_t)h * D_IN * H_D);
            unsigned int* dst32 = (unsigned int*)((p == 0) ? wqs : (p == 1) ? wks : wvs);
#pragma unroll
            for (int i = 0; i < 4; i++) {
                int e2 = t + 256 * i;            // pair index, 0..1023
                int n = e2 & 63, kp = e2 >> 6;   // col n, k-pair
                float w0 = W[(size_t)(kk + 2 * kp)     * H_D + n];
                float w1 = W[(size_t)(kk + 2 * kp + 1) * H_D + n];
                dst32[n * (K_TILE / 2) + kp] = pack_bf16x2(w0, w1);
            }
        }
        __syncthreads();

        // A fragment: 16 rows of x, K=0..31
        const int mrow = w * 16 + mloc;
        BFrag a;
        {
            const uint4* rp = (const uint4*)&xs[mrow * K_TILE];
            a.q[0] = rp[qsel];
            a.q[1] = rp[2 + qsel];
        }
#pragma unroll
        for (int p = 0; p < 3; p++) {
            const unsigned short* wl = (p == 0) ? wqs : (p == 1) ? wks : wvs;
            // batch the 4 B-fragment loads, then issue 4 WMMAs (1 wait / 4 wmma)
            BFrag bf[4];
#pragma unroll
            for (int nt = 0; nt < 4; nt++) {
                int n = nt * 16 + mloc;
                const uint4* rp = (const uint4*)&wl[n * K_TILE];
                bf[nt].q[0] = rp[qsel];
                bf[nt].q[1] = rp[2 + qsel];
            }
#pragma unroll
            for (int nt = 0; nt < 4; nt++) {
                acc[p][nt] = __builtin_amdgcn_wmma_f32_16x16x32_bf16(
                    false, a.v, false, bf[nt].v, (short)0, acc[p][nt], false, false);
            }
        }
        __syncthreads();
    }

    // bias + store (q,k row-major bf16 ; v transposed bf16)
    const int roff = (lane < 16) ? 0 : 8;
#pragma unroll
    for (int nt = 0; nt < 4; nt++) {
        int n = nt * 16 + mloc;
        float bqv = bq[h * H_D + n];
        float bkv = bk[h * H_D + n];
        float bvv = bv[h * H_D + n];
#pragma unroll
        for (int j = 0; j < 8; j++) {
            int r = rbase + w * 16 + j + roff;   // global row in [0, B*S)
            int b = r >> 11;                     // / S_LEN
            int s = r & (S_LEN - 1);
            size_t rowix = (size_t)(b * N_H + h) * S_LEN + s;
            qws[rowix * H_D + n] = f32_to_bf16(acc[0][nt][j] + bqv);
            kws[rowix * H_D + n] = f32_to_bf16(acc[1][nt][j] + bkv);
            vtws[((size_t)(b * N_H + h) * H_D + n) * S_LEN + s] =
                f32_to_bf16(acc[2][nt][j] + bvv);
        }
    }
}

// ---------------------------------------------------------------------------
// Kernel 2: flash-style attention with online softmax (bf16 WMMA)
// out[b, s, h*64 + n] = (softmax(q k^T) / sqrt(64)) @ v
// ---------------------------------------------------------------------------
__global__ __launch_bounds__(256) void attn_kernel(
    const unsigned short* __restrict__ qws,
    const unsigned short* __restrict__ kws,
    const unsigned short* __restrict__ vtws,
    float* __restrict__ out)
{
    __shared__ __align__(16) unsigned short pbuf[8][16 * 64]; // per-wave p tile

    const int qt   = blockIdx.x & 15;        // query tile (128 rows)
    const int bh   = blockIdx.x >> 4;        // b*N_H + h
    const int t    = threadIdx.x;
    const int lane = t & 31;
    const int w    = t >> 5;
    const int mloc = lane & 15;
    const int kb   = (lane < 16) ? 0 : 8;
    const int qsel = kb >> 3;
    const size_t rowbase = (size_t)bh * S_LEN;

    // q fragments for this wave's 16 rows (K = 64, two chunks) kept in regs
    const int qrow = qt * 128 + w * 16 + mloc;
    BFrag qf[2];
#pragma unroll
    for (int kc = 0; kc < 2; kc++) {
        const uint4* rp = (const uint4*)&qws[(rowbase + qrow) * H_D + kc * 32];
        qf[kc].q[0] = rp[qsel];
        qf[kc].q[1] = rp[2 + qsel];
    }

    float mrun[8], lrun[8];
    v8f o[4];
#pragma unroll
    for (int j = 0; j < 8; j++) { mrun[j] = -__builtin_inff(); lrun[j] = 0.0f; }
#pragma unroll
    for (int nt = 0; nt < 4; nt++)
#pragma unroll
        for (int j = 0; j < 8; j++) o[nt][j] = 0.0f;

    unsigned short* pw = pbuf[w];

    for (int kt = 0; kt < S_LEN; kt += 64) {
        // keep the next key/value tile warm in cache (gfx1250 global_prefetch)
        if (kt + 64 < S_LEN) {
            __builtin_prefetch(&kws[(rowbase + kt + 64 + mloc * 4) * H_D], 0, 1);
            __builtin_prefetch(&vtws[((size_t)bh * H_D + mloc * 4) * S_LEN + kt + 64], 0, 1);
        }

        // ---- scores s = q k^T for 64 keys (NO pre-softmax scale, per ref) ----
        v8f s4[4];
#pragma unroll
        for (int nt = 0; nt < 4; nt++)
#pragma unroll
            for (int j = 0; j < 8; j++) s4[nt][j] = 0.0f;

#pragma unroll
        for (int kc = 0; kc < 2; kc++) {
            BFrag kf[4];
#pragma unroll
            for (int nt = 0; nt < 4; nt++) {
                int trow = kt + nt * 16 + mloc;
                const uint4* rp = (const uint4*)&kws[(rowbase + trow) * H_D + kc * 32];
                kf[nt].q[0] = rp[qsel];
                kf[nt].q[1] = rp[2 + qsel];
            }
#pragma unroll
            for (int nt = 0; nt < 4; nt++) {
                s4[nt] = __builtin_amdgcn_wmma_f32_16x16x32_bf16(
                    false, qf[kc].v, false, kf[nt].v, (short)0, s4[nt], false, false);
            }
        }

        // ---- online softmax update (reductions within 16-lane halves) ----
        float mnew[8], scale[8];
#pragma unroll
        for (int j = 0; j < 8; j++) {
            float v0 = fmaxf(fmaxf(s4[0][j], s4[1][j]), fmaxf(s4[2][j], s4[3][j]));
            v0 = fmaxf(v0, __shfl_xor(v0, 1, 32));
            v0 = fmaxf(v0, __shfl_xor(v0, 2, 32));
            v0 = fmaxf(v0, __shfl_xor(v0, 4, 32));
            v0 = fmaxf(v0, __shfl_xor(v0, 8, 32));
            mnew[j]  = fmaxf(mrun[j], v0);
            scale[j] = __expf(mrun[j] - mnew[j]);
            mrun[j]  = mnew[j];
        }
        const int prow_off = (lane < 16) ? 0 : 8;
#pragma unroll
        for (int nt = 0; nt < 4; nt++) {
#pragma unroll
            for (int j = 0; j < 8; j++) {
                float p = __expf(s4[nt][j] - mnew[j]);
                s4[nt][j] = p;
                pw[(j + prow_off) * 64 + nt * 16 + mloc] = f32_to_bf16(p);
            }
        }
#pragma unroll
        for (int j = 0; j < 8; j++) {
            float r = (s4[0][j] + s4[1][j]) + (s4[2][j] + s4[3][j]);
            r += __shfl_xor(r, 1, 32);
            r += __shfl_xor(r, 2, 32);
            r += __shfl_xor(r, 4, 32);
            r += __shfl_xor(r, 8, 32);
            lrun[j] = lrun[j] * scale[j] + r;
#pragma unroll
            for (int nt = 0; nt < 4; nt++) o[nt][j] *= scale[j];
        }

        // ensure p tile stores complete before fragment re-read (wave-private)
        asm volatile("s_wait_dscnt 0x0" ::: "memory");

        // ---- o += p @ v (K = 64 keys, two chunks) ----
        BFrag pf[2];
#pragma unroll
        for (int pc = 0; pc < 2; pc++) {
            const uint4* rp = (const uint4*)&pw[mloc * 64 + pc * 32];
            pf[pc].q[0] = rp[qsel];
            pf[pc].q[1] = rp[2 + qsel];
        }
#pragma unroll
        for (int pc = 0; pc < 2; pc++) {
            BFrag vf[4];
#pragma unroll
            for (int nt = 0; nt < 4; nt++) {
                int n = nt * 16 + mloc;
                const uint4* rp = (const uint4*)&vtws[((size_t)bh * H_D + n) * S_LEN + kt + pc * 32];
                vf[nt].q[0] = rp[qsel];
                vf[nt].q[1] = rp[2 + qsel];
            }
#pragma unroll
            for (int nt = 0; nt < 4; nt++) {
                o[nt] = __builtin_amdgcn_wmma_f32_16x16x32_bf16(
                    false, pf[pc].v, false, vf[nt].v, (short)0, o[nt], false, false);
            }
        }
    }

    // ---- normalize (softmax sum) * 1/sqrt(64) and write out ----
    const int bb = bh >> 4, hh = bh & 15;
#pragma unroll
    for (int j = 0; j < 8; j++) {
        float invl = 0.125f / lrun[j];
        int srow = qt * 128 + w * 16 + j + ((lane < 16) ? 0 : 8);
        float* orow = out + ((size_t)bb * S_LEN + srow) * (N_H * H_D) + hh * H_D;
#pragma unroll
        for (int nt = 0; nt < 4; nt++)
            orow[nt * 16 + mloc] = o[nt][j] * invl;
    }
}

extern "C" void kernel_launch(void* const* d_in, const int* in_sizes, int n_in,
                              void* d_out, int out_size, void* d_ws, size_t ws_size,
                              hipStream_t stream) {
    const float* x  = (const float*)d_in[0];
    const float* Wq = (const float*)d_in[1];
    const float* bq = (const float*)d_in[2];
    const float* Wk = (const float*)d_in[3];
    const float* bk = (const float*)d_in[4];
    const float* Wv = (const float*)d_in[5];
    const float* bv = (const float*)d_in[6];
    float* out = (float*)d_out;

    // workspace: q, k (row-major) and v^T, all bf16, 8 MB each = 24 MB total
    const size_t elems = (size_t)B_SZ * N_H * S_LEN * H_D;
    unsigned short* qws  = (unsigned short*)d_ws;
    unsigned short* kws  = qws + elems;
    unsigned short* vtws = kws + elems;

    qkv_proj_kernel<<<dim3(N_H * (B_SZ * S_LEN / M_TILE)), 256, 0, stream>>>(
        x, Wq, bq, Wk, bk, Wv, bv, qws, kws, vtws);
    attn_kernel<<<dim3(B_SZ * N_H * (S_LEN / 128)), 256, 0, stream>>>(
        qws, kws, vtws, out);
}
